// StgcnLayer_18760417148961
// MI455X (gfx1250) — compile-verified, matrix operand order checked by reading, not run on hardware
//
#include <hip/hip_runtime.h>
#include <hip/hip_bf16.h>

// ---------------- problem constants ----------------
#define NB    8
#define CIN   128
#define COUT  128
#define LDIM  2048
#define VDIM  25
#define PDIM  3
#define KWIN  9

#define LT    16            // l rows per K1 workgroup
#define JCOLS (LT * VDIM)   // 400 GEMM columns per workgroup
#define XS_STRIDE 136       // halves per j row in LDS (128 cin + 8 pad)
#define Y_STRIDE  520       // halves per (p,c) row in LDS (16 l * 32 w + 8 pad)
#define Y_HOFF    (JCOLS * XS_STRIDE)           // 54400 halves
#define LDS_BYTES ((Y_HOFF + 192 * Y_STRIDE) * 2)  // 308480 B <= 320KB WGP LDS

typedef __bf16 v8bf  __attribute__((ext_vector_type(8)));
typedef __bf16 v16bf __attribute__((ext_vector_type(16)));
typedef float  v8f   __attribute__((ext_vector_type(8)));

// Load a 16-half WMMA operand fragment as two 16B chunks (K and K+16 halves).
__device__ __forceinline__ v16bf load_frag(const __bf16* p) {
  v8bf lo = *(const v8bf*)(p);
  v8bf hi = *(const v8bf*)(p + 16);
  v16bf r;
#pragma unroll
  for (int i = 0; i < 8; ++i) { r[i] = lo[i]; r[i + 8] = hi[i]; }
  return r;
}

// ---------------- K0: weight prep (fp32 -> bf16) + stats zero ----------------
__global__ __launch_bounds__(256) void k0_prep(
    const float* __restrict__ conv_w, const float* __restrict__ A,
    const float* __restrict__ edge, __bf16* __restrict__ Wb,
    __bf16* __restrict__ AwB, float* __restrict__ stats)
{
  int idx = blockIdx.x * blockDim.x + threadIdx.x;
  int stride = gridDim.x * blockDim.x;
  // Wb[o][cin] = bf16(conv_w[o][cin]),   o = p*128 + c
  for (int i = idx; i < PDIM * COUT * CIN; i += stride)
    Wb[i] = (__bf16)conv_w[i];
  // Per-lane-ordered B-fragments of Aw = A (*) edge, padded 25->32, zero filled.
  // i = ((p*2 + vt)*32 + lane)*16 + h
  for (int i = idx; i < PDIM * 2 * 32 * 16; i += stride) {
    int h    = i & 15;
    int lane = (i >> 4) & 31;
    int vt   = (i >> 9) & 1;
    int p    = i >> 10;
    int w = (lane < 16) ? ((h < 8) ? h : h + 8) : ((h < 8) ? h + 8 : h + 16);
    int v = vt * 16 + (lane & 15);
    float val = 0.f;
    if (w < VDIM && v < VDIM) val = A[(p * VDIM + w) * VDIM + v] * edge[w * VDIM + v];
    AwB[i] = (__bf16)val;
  }
  if (idx < 512) stats[idx] = 0.f;  // sum / sumsq / scale / shift (128 each)
}

// ---------------- K1: fused 1x1 conv GEMM + graph aggregation (WMMA) --------
// grid = (L/16, N, 2 channel-halves), block = 256 (8 wave32)
__global__ __launch_bounds__(256) void k1_gemm_graph(
    const float* __restrict__ x, const __bf16* __restrict__ Wb,
    const __bf16* __restrict__ AwB, float* __restrict__ z)
{
  extern __shared__ char smem_raw[];
  __bf16* xs = (__bf16*)smem_raw;   // [400 j][136]  bf16 x^T tile
  __bf16* Yl = xs + Y_HOFF;         // [192 (p,c)][520] bf16 conv output tile

  const int l0  = blockIdx.x * LT;
  const int n   = blockIdx.y;
  const int ch  = blockIdx.z;       // channel half: c in [ch*64, ch*64+64)
  const int tid = threadIdx.x;

  // ---- phase 0: stage x tile (fp32 -> bf16, transposed to [j][cin]) ----
  const float* xsrc = x + (size_t)n * CIN * LDIM * VDIM + (size_t)l0 * VDIM;
  for (int e = tid; e < CIN * JCOLS; e += 256) {
    int cin = e / JCOLS;
    int j   = e - cin * JCOLS;      // j = l_local*25 + v (contiguous in global)
    xs[j * XS_STRIDE + cin] = (__bf16)xsrc[(size_t)cin * LDIM * VDIM + j];
  }
  // zero Y's K-padding (w = 25..31) so stage-2 K=32 WMMA sums zeros there
  for (int e = tid; e < 192 * LT * 7; e += 256) {
    int row = e / (LT * 7);
    int rem = e - row * (LT * 7);
    int l   = rem / 7;
    int w   = 25 + (rem - l * 7);
    Yl[row * Y_STRIDE + l * 32 + w] = (__bf16)0.0f;
  }
  __syncthreads();

  const int wv   = tid >> 5;
  const int lane = tid & 31;
  const int mrow = lane & 15;                 // per-lane matrix row / column
  const int koff = (lane & 16) ? 8 : 0;       // per-lane K chunk select
  const int hi8  = (lane & 16) ? 8 : 0;       // D rows r+8 for upper half-wave

  // ---- phase 1: Y[o][j] = sum_cin W[o][cin] * x[cin][j]  (K=128, 4 WMMAs) ----
  // 12 o-tiles (3p x 4 c-subtiles) x 25 j-tiles = 300 C-tiles over 8 waves
  for (int t = wv; t < 12 * 25; t += 8) {
    int mt = t / 25;
    int jt = t - mt * 25;
    int p  = mt >> 2;
    int cq = mt & 3;
    const __bf16* wrow = Wb + (size_t)(p * COUT + ch * 64 + cq * 16 + mrow) * CIN;
    int j = jt * 16 + mrow;                   // lane's B column == lane's D column
    const __bf16* brow = xs + j * XS_STRIDE;
    v8f acc = {};
#pragma unroll
    for (int ks = 0; ks < 4; ++ks) {
      v16bf af = load_frag(wrow + ks * 32 + koff);
      v16bf bf = load_frag(brow + ks * 32 + koff);
      acc = __builtin_amdgcn_wmma_f32_16x16x32_bf16(
          false, af, false, bf, (short)0, acc, false, false);
    }
    int l = j / 25, w = j - l * 25;           // column j -> (l, w) in Y
    int cb = cq * 16 + hi8;
#pragma unroll
    for (int r = 0; r < 8; ++r)
      Yl[(p * 64 + cb + r) * Y_STRIDE + l * 32 + w] = (__bf16)acc[r];
  }
  __syncthreads();

  // ---- phase 2: z[c][l][v] = sum_p sum_w Y_p[c][l][w] * Aw[p][w][v] ----
  v16bf bAw[PDIM][2];
#pragma unroll
  for (int p = 0; p < PDIM; ++p)
#pragma unroll
    for (int vt = 0; vt < 2; ++vt)
      bAw[p][vt] = *(const v16bf*)(AwB + (((p * 2 + vt) * 32) + lane) * 16);

  float* zbase = z + (size_t)n * COUT * LDIM * VDIM;
  for (int s = wv; s < 128; s += 8) {         // 4 c-tiles x 16 l x 2 v-tiles
    int ct = s >> 5;
    int l  = (s >> 1) & 15;
    int vt = s & 1;
    int crow = ct * 16 + mrow;
    v8f acc = {};
#pragma unroll
    for (int p = 0; p < PDIM; ++p) {
      v16bf af = load_frag(Yl + (p * 64 + crow) * Y_STRIDE + l * 32 + koff);
      acc = __builtin_amdgcn_wmma_f32_16x16x32_bf16(
          false, af, false, bAw[p][vt], (short)0, acc, false, false);
    }
    int v = vt * 16 + (lane & 15);
    if (v < VDIM) {
#pragma unroll
      for (int r = 0; r < 8; ++r) {
        int cg = ch * 64 + ct * 16 + hi8 + r;
        zbase[((size_t)cg * LDIM + (l0 + l)) * VDIM + v] = acc[r];
      }
    }
  }
}

// ---------------- K2: causal 9-tap temporal sum + BN partial stats ----------
// grid = N*COUT blocks (one (n,c) plane each), block = 256.
// Thread t owns column v = t&31 (active if v < 25) over l-strip t>>5.
// Sliding window kept incrementally: 2 coalesced loads + 2 adds per output
// (z[l-9] re-read is always L0-hot), instead of 9 loads per output.
#define LSTRIP (LDIM / 8)   // 256 l's per thread strip
__global__ __launch_bounds__(256) void k2_window_stats(
    const float* __restrict__ z, float* __restrict__ outp, float* __restrict__ stats)
{
  __shared__ float red1[256], red2[256];
  const int nc = blockIdx.x;
  const int c  = nc & (COUT - 1);
  const float* zp = z + (size_t)nc * LDIM * VDIM;
  float* op = outp + (size_t)nc * LDIM * VDIM;

  const int vv    = threadIdx.x & 31;
  const int strip = threadIdx.x >> 5;
  const bool act  = (vv < VDIM);
  const int l0    = strip * LSTRIP;

  float s1 = 0.f, s2 = 0.f;
  float S = 0.f;
  if (act) {
    // warm up over the 8 halo rows, then stream the strip
    for (int l = l0 - (KWIN - 1); l < l0 + LSTRIP; ++l) {
      if (l >= 0) S += zp[l * VDIM + vv];
      int lm = l - KWIN;
      // lm >= l0-8 keeps only terms added during this strip's warm-up;
      // for strip 0 (l0 == 0) this reduces to lm >= 0 == reference zero-pad.
      if (lm >= 0 && lm >= l0 - (KWIN - 1)) S -= zp[lm * VDIM + vv];
      if (l >= l0) {
        op[l * VDIM + vv] = S;
        s1 += S;
        s2 += S * S;
      }
    }
  }
  red1[threadIdx.x] = s1;
  red2[threadIdx.x] = s2;
  __syncthreads();
  for (int off = 128; off > 0; off >>= 1) {
    if (threadIdx.x < off) {
      red1[threadIdx.x] += red1[threadIdx.x + off];
      red2[threadIdx.x] += red2[threadIdx.x + off];
    }
    __syncthreads();
  }
  if (threadIdx.x == 0) {
    atomicAdd(&stats[c], red1[0]);
    atomicAdd(&stats[COUT + c], red2[0]);
  }
}

// ---------------- K3: per-channel scale/shift ----------------
__global__ void k3_finalize(const float* __restrict__ gamma,
                            const float* __restrict__ beta, float* __restrict__ stats)
{
  int c = threadIdx.x;
  if (c < COUT) {
    const float cnt = (float)(NB * LDIM * VDIM);   // 409600
    float mean = stats[c] / cnt;
    float var  = stats[COUT + c] / cnt - mean * mean;  // biased, matches jnp.var
    float rstd = rsqrtf(var + 1e-5f);
    float sc = gamma[c] * rstd;
    stats[2 * COUT + c] = sc;
    stats[3 * COUT + c] = beta[c] - mean * sc;
  }
}

// ---------------- K4: BN affine + ReLU + residual + ReLU (in place) ---------
__global__ __launch_bounds__(256) void k4_bn_relu_res(
    const float* __restrict__ x, const float* __restrict__ stats,
    float* __restrict__ out)
{
  size_t i4 = (size_t)blockIdx.x * 256 + threadIdx.x;   // float4 index
  int c = (int)((i4 / (LDIM * VDIM / 4)) & (COUT - 1)); // uniform within float4
  float sc = stats[2 * COUT + c], sh = stats[3 * COUT + c];
  float4 pv = ((const float4*)out)[i4];
  float4 xv = ((const float4*)x)[i4];
  float4 r;
  r.x = fmaxf(fmaxf(pv.x * sc + sh, 0.f) + xv.x, 0.f);
  r.y = fmaxf(fmaxf(pv.y * sc + sh, 0.f) + xv.y, 0.f);
  r.z = fmaxf(fmaxf(pv.z * sc + sh, 0.f) + xv.z, 0.f);
  r.w = fmaxf(fmaxf(pv.w * sc + sh, 0.f) + xv.w, 0.f);
  ((float4*)out)[i4] = r;
}

// ---------------- launch ----------------
extern "C" void kernel_launch(void* const* d_in, const int* in_sizes, int n_in,
                              void* d_out, int out_size, void* d_ws, size_t ws_size,
                              hipStream_t stream) {
  (void)in_sizes; (void)n_in; (void)out_size; (void)ws_size;
  const float* x      = (const float*)d_in[0];
  const float* A      = (const float*)d_in[1];
  const float* edge   = (const float*)d_in[2];
  const float* conv_w = (const float*)d_in[3];
  const float* gamma  = (const float*)d_in[4];
  const float* beta   = (const float*)d_in[5];
  float* out = (float*)d_out;

  char* ws = (char*)d_ws;
  __bf16* Wb   = (__bf16*)(ws);             // 384*128 bf16   = 98304 B
  __bf16* AwB  = (__bf16*)(ws + 98304);     // 3*2*32*16 bf16 = 6144 B
  float*  stats = (float*)(ws + 104448);    // 512 f32
  float*  z    = (float*)(ws + (1 << 20));  // 8*128*2048*25 f32 = 209.7 MB

  k0_prep<<<192, 256, 0, stream>>>(conv_w, A, edge, Wb, AwB, stats);
  dim3 g1(LDIM / LT, NB, 2);
  k1_gemm_graph<<<g1, 256, LDS_BYTES, stream>>>(x, Wb, AwB, z);
  k2_window_stats<<<NB * COUT, 256, 0, stream>>>(z, out, stats);
  k3_finalize<<<1, COUT, 0, stream>>>(gamma, beta, stats);
  int total4 = NB * COUT * LDIM * VDIM / 4;
  k4_bn_relu_res<<<total4 / 256, 256, 0, stream>>>(x, stats, out);
}
